// AttentionLayer_19765439496635
// MI455X (gfx1250) — matrix operand order repaired and checked by reading, not compiled, not run on hardware
//
#include <hip/hip_runtime.h>

// ---------------------------------------------------------------------------
// Attention layer for MI455X (gfx1250, wave32, WMMA).
// All GEMMs run through v_wmma_f32_16x16x32_f16 with f32 accumulation.
// Flash-attention (online softmax) avoids materializing the s x s scores.
// Workspace need: ~32 MB (Q,K f16 [b,h,s,d]; V^T f16 [b,h,d,s]; attn f16; mask f32).
// ---------------------------------------------------------------------------

typedef _Float16 half_t;
typedef __attribute__((ext_vector_type(16))) _Float16 v16h;
typedef __attribute__((ext_vector_type(8)))  _Float16 v8h;
typedef __attribute__((ext_vector_type(8)))  float    v8f;
typedef __attribute__((ext_vector_type(4)))  float    v4f;

#define D_MODEL  1024
#define N_HEADS  16
#define HEAD_DIM 64
#define BATCH    2
#define SEQ      2048
#define MROWS    (BATCH * SEQ)
#define BK       64

union V16U { v16h v; v8h h[2]; };

static __device__ inline v16h combine16(v8h lo, v8h hi) {
  V16U u; u.h[0] = lo; u.h[1] = hi; return u.v;
}

// ---------------------------------------------------------------------------
// Pad-mask precompute: maskf[b*SEQ+s] = (ids != 0) ? 0 : -1e9
// ---------------------------------------------------------------------------
__global__ void mask_kernel(const int* __restrict__ ids, float* __restrict__ maskf) {
  int i = blockIdx.x * blockDim.x + threadIdx.x;
  if (i < BATCH * SEQ) maskf[i] = (ids[i] != 0) ? 0.0f : -1e9f;
}

// ---------------------------------------------------------------------------
// Projection GEMM:  Y = X @ W^T + b   (M=4096, N=1024, K=1024)
// Block tile 64(M) x 128(N), K-step 64; 8 waves, each computes 32x32.
// VT_MODE==0 : store [b, h, s, d]   (Q and K)
// VT_MODE==1 : store [b, h, d, s]   (V transposed for contiguous PV B-frags)
// ---------------------------------------------------------------------------
template <int VT_MODE>
__global__ __launch_bounds__(256)
void proj_gemm_f32(const float* __restrict__ X, const float* __restrict__ Wt,
                   const float* __restrict__ bias, half_t* __restrict__ out) {
  __shared__ alignas(32) half_t As[64][BK];
  __shared__ alignas(32) half_t Bs[128][BK];

  const int tid  = threadIdx.x;
  const int lane = tid & 31;
  const int wave = tid >> 5;
  const int r    = lane & 15;     // row (A) / col (B,C)
  const int hg   = lane >> 4;     // half-group of the wave
  const int wr   = wave >> 2;     // wave M-tile (0..1)
  const int wc   = wave & 3;      // wave N-tile (0..3)
  const int m0   = blockIdx.x * 64;
  const int n0   = blockIdx.y * 128;

  // staging assignments (BK=64): A 16 elems/thread, B 32 elems/thread
  const int rowA = tid >> 2, ka = (tid & 3) * 16;
  const int rowB = tid >> 1, kb = (tid & 1) * 32;
  const float* srcA = X  + (size_t)(m0 + rowA) * D_MODEL + ka;
  const float* srcB = Wt + (size_t)(n0 + rowB) * D_MODEL + kb;

  v8f acc[2][2] = {};

  for (int kk = 0; kk < D_MODEL; kk += BK) {
    if (kk + BK < D_MODEL) {               // prefetch next K-tile (gfx1250 global_prefetch_b8)
      __builtin_prefetch(srcA + kk + BK, 0, 3);
      __builtin_prefetch(srcB + kk + BK, 0, 3);
    }
    {
      half_t* dst = &As[rowA][ka];
      #pragma unroll
      for (int j = 0; j < 4; j++) {
        v4f x = *(const v4f*)(srcA + kk + 4 * j);
        #pragma unroll
        for (int i = 0; i < 4; i++) dst[4 * j + i] = (half_t)x[i];
      }
    }
    {
      half_t* dst = &Bs[rowB][kb];
      #pragma unroll
      for (int j = 0; j < 8; j++) {
        v4f x = *(const v4f*)(srcB + kk + 4 * j);
        #pragma unroll
        for (int i = 0; i < 4; i++) dst[4 * j + i] = (half_t)x[i];
      }
    }
    __syncthreads();

    v16h af[2][2], bf[2][2];
    #pragma unroll
    for (int kf = 0; kf < 2; kf++) {
      #pragma unroll
      for (int mi = 0; mi < 2; mi++) {
        const half_t* a = &As[wr * 32 + mi * 16 + r][kf * 32];
        af[mi][kf] = *(const v16h*)(a + hg * 16);
      }
      #pragma unroll
      for (int ni = 0; ni < 2; ni++)
        bf[ni][kf] = *(const v16h*)&Bs[wc * 32 + ni * 16 + r][kf * 32 + hg * 16];
    }
    #pragma unroll
    for (int kf = 0; kf < 2; kf++)
      #pragma unroll
      for (int mi = 0; mi < 2; mi++)
        #pragma unroll
        for (int ni = 0; ni < 2; ni++)
          acc[mi][ni] = __builtin_amdgcn_wmma_f32_16x16x32_f16(
              false, af[mi][kf], false, bf[ni][kf], (short)0, acc[mi][ni], false, false);
    __syncthreads();
  }

  // epilogue: C layout row = g + 8*hg, col = r   (no branches: VT_MODE compile-time)
  const int bidx  = m0 >> 11;
  const int sbase = (m0 & 2047) + wr * 32;
  float bv[2];
  int   hh[2], dd[2];
  #pragma unroll
  for (int ni = 0; ni < 2; ni++) {
    int n = n0 + wc * 32 + ni * 16 + r;
    bv[ni] = bias[n];
    hh[ni] = n >> 6;
    dd[ni] = n & 63;
  }
  #pragma unroll
  for (int mi = 0; mi < 2; mi++)
    #pragma unroll
    for (int ni = 0; ni < 2; ni++)
      #pragma unroll
      for (int g = 0; g < 8; g++) {
        int s = sbase + mi * 16 + g + 8 * hg;
        float val = acc[mi][ni][g] + bv[ni];
        size_t idx = (VT_MODE == 0)
            ? ((((size_t)bidx * N_HEADS + hh[ni]) * SEQ + s) * HEAD_DIM + dd[ni])
            : ((((size_t)bidx * N_HEADS + hh[ni]) * HEAD_DIM + dd[ni]) * SEQ + s);
        out[idx] = (half_t)val;
      }
}

// ---------------------------------------------------------------------------
// One 32-key flash-attention chunk. CAUSAL=false for chunks fully below the
// diagonal (the hot path: no compares/cndmasks at all).
// ---------------------------------------------------------------------------
template <bool CAUSAL>
__device__ __forceinline__ void attn_chunk(
    int kc0, int q0, int r, int hg, half_t (*PsW)[32],
    const half_t* __restrict__ Kb, const half_t* __restrict__ Vb,
    const float* __restrict__ mb, const v16h (&qf)[2],
    float (&mrun)[8], float (&lrun)[8], v8f (&o)[4]) {
  const float scale = 0.125f; // 1/sqrt(64)

  // S = Q K^T over this 32-key chunk: two 16x16 tiles
  v8f sacc[2] = {};
  #pragma unroll
  for (int t = 0; t < 2; t++)
    #pragma unroll
    for (int kf = 0; kf < 2; kf++) {
      const half_t* kbp = Kb + (size_t)(kc0 + t * 16 + r) * HEAD_DIM + kf * 32 + hg * 16;
      v16h bfr = *(const v16h*)kbp;
      sacc[t] = __builtin_amdgcn_wmma_f32_16x16x32_f16(
          false, qf[kf], false, bfr, (short)0, sacc[t], false, false);
    }

  // pad-mask values: one load per key-tile per lane
  float mk[2];
  #pragma unroll
  for (int t = 0; t < 2; t++) mk[t] = mb[kc0 + t * 16 + r];

  // mask + online softmax (C layout: row = g+8*hg, col = r)
  float p[2][8];
  #pragma unroll
  for (int g = 0; g < 8; g++) {
    float m_t = -3e38f;
    #pragma unroll
    for (int t = 0; t < 2; t++) {
      float v = sacc[t][g] * scale + mk[t];
      if (CAUSAL) {
        int qrow = q0 + g + 8 * hg;
        if (kc0 + t * 16 + r > qrow) v = -1e9f;
      }
      p[t][g] = v;
      m_t = fmaxf(m_t, v);
    }
    #pragma unroll
    for (int off = 1; off < 16; off <<= 1)
      m_t = fmaxf(m_t, __shfl_xor(m_t, off, 32));
    float mnew = fmaxf(mrun[g], m_t);
    float corr = __expf(mrun[g] - mnew);
    mrun[g] = mnew;
    float rs = 0.0f;
    #pragma unroll
    for (int t = 0; t < 2; t++) {
      float e = __expf(p[t][g] - mnew);
      p[t][g] = e;
      rs += e;
    }
    #pragma unroll
    for (int off = 1; off < 16; off <<= 1) rs += __shfl_xor(rs, off, 32);
    lrun[g] = lrun[g] * corr + rs;
    #pragma unroll
    for (int nt = 0; nt < 4; nt++) o[nt][g] *= corr;
  }

  // P (C-frag f32) -> LDS -> A-frag f16
  #pragma unroll
  for (int t = 0; t < 2; t++)
    #pragma unroll
    for (int g = 0; g < 8; g++)
      PsW[g + 8 * hg][t * 16 + r] = (half_t)p[t][g];
  __syncthreads();

  const half_t* pb = &PsW[r][0];
  v16h pf = *(const v16h*)(pb + hg * 16);

  // O += P V ; V^T is [d, s] so B-fragments are contiguous 32B loads
  #pragma unroll
  for (int nt = 0; nt < 4; nt++) {
    const half_t* vb = Vb + (size_t)(nt * 16 + r) * SEQ + kc0 + hg * 16;
    v16h vf = *(const v16h*)vb;
    o[nt] = __builtin_amdgcn_wmma_f32_16x16x32_f16(
        false, pf, false, vf, (short)0, o[nt], false, false);
  }
  __syncthreads();
}

// ---------------------------------------------------------------------------
// Flash attention per (b,h): grid = (SEQ/64, BATCH*N_HEADS), 128 threads.
// Each wave owns 16 query rows; key chunks of 32; split into a no-causal
// main loop and a short causal tail (per-wave split; each chunk executes the
// same two barriers, so barrier sequences stay aligned across waves).
// ---------------------------------------------------------------------------
__global__ __launch_bounds__(128)
void flash_attn(const half_t* __restrict__ Qh, const half_t* __restrict__ Kh,
                const half_t* __restrict__ Vt, const float* __restrict__ maskf,
                half_t* __restrict__ attnb) {
  __shared__ alignas(32) half_t Ps[4][16][32];

  const int tid  = threadIdx.x;
  const int wave = tid >> 5;
  const int lane = tid & 31;
  const int r    = lane & 15;
  const int hg   = lane >> 4;
  const int bh   = blockIdx.y;
  const int b    = bh >> 4;
  const int h    = bh & 15;
  const int q0   = blockIdx.x * 64 + wave * 16;

  const half_t* Qb = Qh + (size_t)bh * SEQ * HEAD_DIM;
  const half_t* Kb = Kh + (size_t)bh * SEQ * HEAD_DIM;
  const half_t* Vb = Vt + (size_t)bh * HEAD_DIM * SEQ;
  const float*  mb = maskf + (size_t)b * SEQ;

  // Q A-fragments for 16 rows x d=64 (two k=32 fragments)
  v16h qf[2];
  #pragma unroll
  for (int kf = 0; kf < 2; kf++) {
    const half_t* base = Qb + (size_t)(q0 + r) * HEAD_DIM + kf * 32 + hg * 16;
    qf[kf] = *(const v16h*)base;
  }

  float mrun[8], lrun[8];
  #pragma unroll
  for (int g = 0; g < 8; g++) { mrun[g] = -1e30f; lrun[g] = 0.0f; }
  v8f o[4] = {};

  const int nch = 2 * blockIdx.x + 2;  // chunks of 32 keys covering keys <= blockIdx.x*64+63
  const int jcf = (q0 + 1) >> 5;       // chunks fully below the diagonal (no causal)
  for (int jc = 0; jc < jcf; jc++)
    attn_chunk<false>(jc * 32, q0, r, hg, Ps[wave], Kb, Vb, mb, qf, mrun, lrun, o);
  for (int jc = jcf; jc < nch; jc++)
    attn_chunk<true>(jc * 32, q0, r, hg, Ps[wave], Kb, Vb, mb, qf, mrun, lrun, o);

  // normalize and store attention output as f16 [b, s, h*64+d]
  #pragma unroll
  for (int g = 0; g < 8; g++) {
    float inv_l = 1.0f / fmaxf(lrun[g], 1e-20f);
    int s = q0 + g + 8 * hg;
    half_t* dst = attnb + ((size_t)b * SEQ + s) * D_MODEL + h * HEAD_DIM;
    #pragma unroll
    for (int nt = 0; nt < 4; nt++)
      dst[nt * 16 + r] = (half_t)(o[nt][g] * inv_l);
  }
}

// ---------------------------------------------------------------------------
// Output projection: out = attn(f16) @ Wo^T + bo, f32 output to d_out.
// ---------------------------------------------------------------------------
__global__ __launch_bounds__(256)
void out_gemm(const half_t* __restrict__ Ah, const float* __restrict__ Wt,
              const float* __restrict__ bias, float* __restrict__ out) {
  __shared__ alignas(32) half_t As[64][BK];
  __shared__ alignas(32) half_t Bs[128][BK];

  const int tid  = threadIdx.x;
  const int lane = tid & 31;
  const int wave = tid >> 5;
  const int r    = lane & 15;
  const int hg   = lane >> 4;
  const int wr   = wave >> 2;
  const int wc   = wave & 3;
  const int m0   = blockIdx.x * 64;
  const int n0   = blockIdx.y * 128;

  const int rowA = tid >> 2, ka = (tid & 3) * 16;
  const int rowB = tid >> 1, kb = (tid & 1) * 32;
  const half_t* srcA = Ah + (size_t)(m0 + rowA) * D_MODEL + ka;
  const float*  srcB = Wt + (size_t)(n0 + rowB) * D_MODEL + kb;

  v8f acc[2][2] = {};

  for (int kk = 0; kk < D_MODEL; kk += BK) {
    if (kk + BK < D_MODEL) {
      __builtin_prefetch(srcA + kk + BK, 0, 3);
      __builtin_prefetch(srcB + kk + BK, 0, 3);
    }
    {
      *(v8h*)&As[rowA][ka]     = *(const v8h*)(srcA + kk);
      *(v8h*)&As[rowA][ka + 8] = *(const v8h*)(srcA + kk + 8);
    }
    {
      half_t* dst = &Bs[rowB][kb];
      #pragma unroll
      for (int j = 0; j < 8; j++) {
        v4f x = *(const v4f*)(srcB + kk + 4 * j);
        #pragma unroll
        for (int i = 0; i < 4; i++) dst[4 * j + i] = (half_t)x[i];
      }
    }
    __syncthreads();

    v16h af[2][2], bf[2][2];
    #pragma unroll
    for (int kf = 0; kf < 2; kf++) {
      #pragma unroll
      for (int mi = 0; mi < 2; mi++) {
        const half_t* a = &As[wr * 32 + mi * 16 + r][kf * 32];
        af[mi][kf] = *(const v16h*)(a + hg * 16);
      }
      #pragma unroll
      for (int ni = 0; ni < 2; ni++)
        bf[ni][kf] = *(const v16h*)&Bs[wc * 32 + ni * 16 + r][kf * 32 + hg * 16];
    }
    #pragma unroll
    for (int kf = 0; kf < 2; kf++)
      #pragma unroll
      for (int mi = 0; mi < 2; mi++)
        #pragma unroll
        for (int ni = 0; ni < 2; ni++)
          acc[mi][ni] = __builtin_amdgcn_wmma_f32_16x16x32_f16(
              false, af[mi][kf], false, bf[ni][kf], (short)0, acc[mi][ni], false, false);
    __syncthreads();
  }

  float bv[2];
  #pragma unroll
  for (int ni = 0; ni < 2; ni++) bv[ni] = bias[n0 + wc * 32 + ni * 16 + r];
  #pragma unroll
  for (int mi = 0; mi < 2; mi++)
    #pragma unroll
    for (int ni = 0; ni < 2; ni++)
      #pragma unroll
      for (int g = 0; g < 8; g++) {
        int m = m0 + wr * 32 + mi * 16 + g + 8 * hg;
        int n = n0 + wc * 32 + ni * 16 + r;
        out[(size_t)m * D_MODEL + n] = acc[mi][ni][g] + bv[ni];
      }
}

// ---------------------------------------------------------------------------
extern "C" void kernel_launch(void* const* d_in, const int* in_sizes, int n_in,
                              void* d_out, int out_size, void* d_ws, size_t ws_size,
                              hipStream_t stream) {
  const float* x   = (const float*)d_in[0];
  const int*   ids = (const int*)d_in[1];
  const float* Wq  = (const float*)d_in[2];
  const float* bq  = (const float*)d_in[3];
  const float* Wk  = (const float*)d_in[4];
  const float* bk  = (const float*)d_in[5];
  const float* Wv  = (const float*)d_in[6];
  const float* bv  = (const float*)d_in[7];
  const float* Wo  = (const float*)d_in[8];
  const float* bo  = (const float*)d_in[9];
  float* out = (float*)d_out;

  // workspace carve-up (~32 MB + 16 KB)
  const size_t per = (size_t)BATCH * N_HEADS * SEQ * HEAD_DIM; // 4M halves
  half_t* Qh    = (half_t*)d_ws;
  half_t* Kh    = Qh + per;
  half_t* Vt    = Kh + per;
  half_t* attnb = Vt + per;                       // [B*S, D] f16
  float*  maskf = (float*)(attnb + (size_t)BATCH * SEQ * D_MODEL);

  mask_kernel<<<(BATCH * SEQ + 255) / 256, 256, 0, stream>>>(ids, maskf);

  dim3 g1(MROWS / 64, D_MODEL / 128);
  proj_gemm_f32<0><<<g1, 256, 0, stream>>>(x, Wq, bq, Qh);
  proj_gemm_f32<0><<<g1, 256, 0, stream>>>(x, Wk, bk, Kh);
  proj_gemm_f32<1><<<g1, 256, 0, stream>>>(x, Wv, bv, Vt);

  dim3 g2(SEQ / 64, BATCH * N_HEADS);
  flash_attn<<<g2, 128, 0, stream>>>(Qh, Kh, Vt, maskf, attnb);

  out_gemm<<<g1, 256, 0, stream>>>(attnb, Wo, bo, out);
}
